// MultiHeadAttention_59176059404468
// MI455X (gfx1250) — matrix operand order repaired and checked by reading, not compiled, not run on hardware
//
#include <hip/hip_runtime.h>

// ---------------------------------------------------------------------------
// MultiHeadAttention forward for MI455X (gfx1250, wave32, WMMA).
// Pipeline: pack fp32->bf16, QKV projections (bf16 WMMA GEMM), causal flash
// attention (bf16 WMMA for QK^T and P*V, fp32 softmax), output projection.
// All LDS tiles use 16B-aligned padded strides so fragments load as
// ds_load_b128; global->LDS staging is software-pipelined through VGPRs.
// Workspace layout (requires ws_size >= 88 MiB):
//   [0,16M)   xb   : x in bf16                  [B*S, 1024]
//   [16,18M)  wqb  : W_q bf16                   [1024,1024]
//   [18,20M)  wkb  : W_k bf16
//   [20,22M)  wvb  : W_v bf16
//   [22,24M)  wob  : W_o bf16
//   [24,40M)  Qb   : Q bf16                     [B,H,S,64]
//   [40,56M)  Kb   : K bf16                     [B,H,S,64]
//   [56,72M)  Vb   : V bf16                     [B,H,S,64]
//   [72,88M)  Ab   : attention output bf16      [B*S, 1024]
// ---------------------------------------------------------------------------

typedef __attribute__((ext_vector_type(16))) __bf16 v16bf;
typedef __attribute__((ext_vector_type(8)))  float  v8f;

#define BATCH  4
#define SEQ    2048
#define DMODEL 1024
#define NHEADS 16
#define DK     64
#define MROWS  (BATCH * SEQ)   // 8192

// LDS row strides (dwords): odd multiples of 4 -> 16B-aligned rows, few bank
// conflicts (stride*row mod 64 distinct over the 16 rows a fragment touches).
#define XS_STR 20   // GEMM X tile   (128 x 16 data dwords)
#define WS_STR 20   // GEMM W tile   (128 x 16 data dwords)
#define KS_STR 36   // flash K tile  (32 x 32 data dwords)
#define VT_STR 20   // flash V^T tile(64 x 16 data dwords)
#define PS_STR 20   // flash P tile  (16 x 16 data dwords, per wave)

union FragAB { v16bf v; unsigned u[8]; };

__device__ __forceinline__ unsigned short f2bf(float f) {
  unsigned u = __builtin_bit_cast(unsigned, f);
  unsigned r = u + 0x7fffu + ((u >> 16) & 1u);   // round-to-nearest-even
  return (unsigned short)(r >> 16);
}

// ---------------------------------------------------------------------------
// fp32 -> bf16 pack
// ---------------------------------------------------------------------------
__global__ __launch_bounds__(256) void pack_bf16_kernel(
    const float* __restrict__ in, unsigned short* __restrict__ out, int n) {
  int i = blockIdx.x * 256 + threadIdx.x;
  if (i < n) out[i] = f2bf(in[i]);
}

// ---------------------------------------------------------------------------
// Y = X @ W^T, X:[MROWS,1024] bf16, W:[1024,1024] bf16 (rows are output cols).
// mode 0: write bf16 into [B,H,S,64] head layout (for Q/K/V).
// mode 1: write fp32 into [MROWS,1024] (final output projection).
// WG = 256 threads = 8 waves (4 M-waves x 2 N-waves). Tile: 128x128, K-step 32.
// Wave computes a 32x64 sub-tile: 2 A-frags x 4 B-frags -> 8 WMMAs / K-step.
// Global->LDS staging is register-pipelined; tile k+64 is prefetched.
// ---------------------------------------------------------------------------
__global__ __launch_bounds__(256) void gemm_xwT_kernel(
    const unsigned* __restrict__ Xd,        // bf16 pairs: [MROWS x 512] dwords
    const unsigned* __restrict__ Wd,        // bf16 pairs: [1024 x 512] dwords
    unsigned short* __restrict__ Yb,        // mode 0 dest
    float* __restrict__ Yf,                 // mode 1 dest
    int mode) {
  __shared__ unsigned Xs[128 * XS_STR];
  __shared__ unsigned Ws[128 * WS_STR];

  const int tid  = threadIdx.x;
  const int w    = tid >> 5;
  const int lane = tid & 31;
  const int l16  = lane & 15;
  const int half = lane >> 4;
  const int wm   = w & 3;        // 0..3  -> M offset wm*32
  const int wn   = w >> 2;       // 0..1  -> N offset wn*64
  const int m0 = blockIdx.y * 128;
  const int n0 = blockIdx.x * 128;

  // A-fragment dword order used here (contiguous k = 16*half + s, B matches): dword(v) = half*8 + v
  int dofs[8];
#pragma unroll
  for (int v = 0; v < 8; ++v) dofs[v] = v + half * 8;

  v8f acc[2][4];
#pragma unroll
  for (int a = 0; a < 2; ++a)
#pragma unroll
    for (int i = 0; i < 4; ++i)
#pragma unroll
      for (int j = 0; j < 8; ++j) acc[a][i][j] = 0.0f;

  // Staging assignment: 2 threads per row, 8 dwords each (both tiles 128x16).
  const int sr = tid >> 1, sc = (tid & 1) * 8;
  const unsigned* xsrc = Xd + (size_t)(m0 + sr) * 512 + sc;
  const unsigned* wsrc = Wd + (size_t)(n0 + sr) * 512 + sc;

  uint4 xr0 = *(const uint4*)(xsrc);
  uint4 xr1 = *(const uint4*)(xsrc + 4);
  uint4 wr0 = *(const uint4*)(wsrc);
  uint4 wr1 = *(const uint4*)(wsrc + 4);

  for (int kt = 0; kt < DMODEL; kt += 32) {
    __syncthreads();
    *(uint4*)&Xs[sr * XS_STR + sc]     = xr0;
    *(uint4*)&Xs[sr * XS_STR + sc + 4] = xr1;
    *(uint4*)&Ws[sr * WS_STR + sc]     = wr0;
    *(uint4*)&Ws[sr * WS_STR + sc + 4] = wr1;
    __syncthreads();

    if (kt + 32 < DMODEL) {            // register-stage next tile during compute
      int kd = (kt + 32) >> 1;
      xr0 = *(const uint4*)(xsrc + kd);
      xr1 = *(const uint4*)(xsrc + kd + 4);
      wr0 = *(const uint4*)(wsrc + kd);
      wr1 = *(const uint4*)(wsrc + kd + 4);
      if (kt + 64 < DMODEL) {          // gfx1250 global_prefetch_b8 two ahead
        __builtin_prefetch(xsrc + ((kt + 64) >> 1), 0, 0);
        __builtin_prefetch(wsrc + ((kt + 64) >> 1), 0, 0);
      }
    }

    FragAB a[2];
#pragma unroll
    for (int ar = 0; ar < 2; ++ar)
#pragma unroll
      for (int v = 0; v < 8; ++v)
        a[ar].u[v] = Xs[(wm * 32 + ar * 16 + l16) * XS_STR + dofs[v]];

#pragma unroll
    for (int nf = 0; nf < 4; ++nf) {
      FragAB b;  // B: col n = W row, sequential k: dwords half*8 .. half*8+7
#pragma unroll
      for (int v = 0; v < 8; ++v)
        b.u[v] = Ws[(wn * 64 + nf * 16 + l16) * WS_STR + half * 8 + v];
      acc[0][nf] = __builtin_amdgcn_wmma_f32_16x16x32_bf16(
          false, a[0].v, false, b.v, (short)0, acc[0][nf], false, false);
      acc[1][nf] = __builtin_amdgcn_wmma_f32_16x16x32_bf16(
          false, a[1].v, false, b.v, (short)0, acc[1][nf], false, false);
    }
  }

  // Epilogue. C layout: n = l16, m = v + 8*half.
#pragma unroll
  for (int ar = 0; ar < 2; ++ar) {
#pragma unroll
    for (int nf = 0; nf < 4; ++nf) {
#pragma unroll
      for (int v = 0; v < 8; ++v) {
        int row = m0 + wm * 32 + ar * 16 + v + 8 * half;
        int col = n0 + wn * 64 + nf * 16 + l16;
        float val = acc[ar][nf][v];
        if (mode == 0) {
          int bsz = row >> 11, s = row & (SEQ - 1);
          int hh = col >> 6, dd = col & 63;
          Yb[((size_t)((bsz * NHEADS + hh) * SEQ + s)) * DK + dd] = f2bf(val);
        } else {
          Yf[(size_t)row * DMODEL + col] = val;
        }
      }
    }
  }
}

// ---------------------------------------------------------------------------
// Causal flash attention. Q,K,V: [B,H,S,64] bf16. Out: [B,S,1024] bf16
// (col = h*64 + d). WG = 256 threads owns 128 query rows of one (b,h);
// wave w owns 16 rows. Key loop in blocks of 32; K/V staged in LDS through
// a register pipeline (V transposed so its B-fragments are contiguous b128s).
// ---------------------------------------------------------------------------
__global__ __launch_bounds__(256) void flash_attn_kernel(
    const unsigned* __restrict__ Qd, const unsigned* __restrict__ Kd,
    const unsigned* __restrict__ Vd, unsigned short* __restrict__ Ob) {
  __shared__ unsigned Ks[32 * KS_STR];
  __shared__ unsigned Vt[64 * VT_STR];
  __shared__ unsigned Ps[8 * 16 * PS_STR];

  const int tid  = threadIdx.x;
  const int w    = tid >> 5;
  const int lane = tid & 31;
  const int l16  = lane & 15;
  const int half = lane >> 4;
  const int qt = blockIdx.x, h = blockIdx.y, b = blockIdx.z;
  const size_t bh = (size_t)(b * NHEADS + h) * SEQ;

  int dofs[8];
#pragma unroll
  for (int v = 0; v < 8; ++v) dofs[v] = v + half * 8;

  // Load Q fragments once (A layout, direct from global, b128-mergeable).
  FragAB q0, q1;
  {
    int qrow = qt * 128 + w * 16 + l16;
    const unsigned* src = Qd + (bh + qrow) * 32;   // 32 dwords per row
#pragma unroll
    for (int v = 0; v < 8; ++v) {
      q0.u[v] = src[dofs[v]];
      q1.u[v] = src[dofs[v] + 16];
    }
  }

  v8f acc[4];
  float Mst[8], Lst[8];
#pragma unroll
  for (int i = 0; i < 4; ++i)
#pragma unroll
    for (int j = 0; j < 8; ++j) acc[i][j] = 0.0f;
#pragma unroll
  for (int v = 0; v < 8; ++v) { Mst[v] = -1e30f; Lst[v] = 0.0f; }

  const float sc = 0.125f * 1.44269504088896340736f;  // 1/sqrt(64) * log2(e)
  const int kEnd = qt * 128 + 128;
  const int wRowMax = qt * 128 + w * 16 + 15;
  unsigned short* Ps16 = (unsigned short*)Ps;
  unsigned short* Vt16 = (unsigned short*)Vt;

  // Staging assignment: 8 threads per key row, 4 dwords each.
  const int kr = tid >> 3, kseg = (tid & 7) * 4;
  const unsigned* ksrc = Kd + (bh + kr) * 32 + kseg;
  const unsigned* vsrc = Vd + (bh + kr) * 32 + kseg;

  uint4 kreg = *(const uint4*)(ksrc);
  uint4 vreg = *(const uint4*)(vsrc);

  for (int kb = 0; kb < kEnd; kb += 32) {
    __syncthreads();
    *(uint4*)&Ks[kr * KS_STR + kseg] = kreg;
    {   // V transposed: Vt16[d][key]
      unsigned p[4] = {vreg.x, vreg.y, vreg.z, vreg.w};
#pragma unroll
      for (int i = 0; i < 4; ++i) {
        int d0 = (kseg + i) * 2;
        Vt16[(d0 + 0) * (2 * VT_STR) + kr] = (unsigned short)(p[i] & 0xffffu);
        Vt16[(d0 + 1) * (2 * VT_STR) + kr] = (unsigned short)(p[i] >> 16);
      }
    }
    __syncthreads();

    if (kb + 32 < kEnd) {              // register-stage next block
      kreg = *(const uint4*)(ksrc + (size_t)(kb + 32) * 32);
      vreg = *(const uint4*)(vsrc + (size_t)(kb + 32) * 32);
    }

    if (kb > wRowMax) continue;        // fully-masked block for this wave

    // ---- scores: S = Q(16x64) @ K^T(64x32) -> two 16x16 C-frags ----
    v8f s01[2];
#pragma unroll
    for (int nh = 0; nh < 2; ++nh) {
      FragAB k0, k1;  // B frags: col n = key, k = d, sequential dwords
#pragma unroll
      for (int v = 0; v < 8; ++v) {
        k0.u[v] = Ks[(nh * 16 + l16) * KS_STR + half * 8 + v];
        k1.u[v] = Ks[(nh * 16 + l16) * KS_STR + 16 + half * 8 + v];
      }
      v8f s;
#pragma unroll
      for (int j = 0; j < 8; ++j) s[j] = 0.0f;
      s = __builtin_amdgcn_wmma_f32_16x16x32_bf16(false, q0.v, false, k0.v,
                                                  (short)0, s, false, false);
      s = __builtin_amdgcn_wmma_f32_16x16x32_bf16(false, q1.v, false, k1.v,
                                                  (short)0, s, false, false);
      s01[nh] = s;
    }

    // ---- scale (base-2) + causal mask ----
#pragma unroll
    for (int nh = 0; nh < 2; ++nh) {
      int keyg = kb + nh * 16 + l16;
#pragma unroll
      for (int v = 0; v < 8; ++v) {
        int qr = qt * 128 + w * 16 + v + 8 * half;
        float sv = s01[nh][v] * sc;
        s01[nh][v] = (keyg <= qr) ? sv : -1e30f;
      }
    }

    // ---- online softmax (row m = v + 8*half; 16-lane shfl reductions) ----
#pragma unroll
    for (int v = 0; v < 8; ++v) {
      float t = fmaxf(s01[0][v], s01[1][v]);
#pragma unroll
      for (int m = 1; m < 16; m <<= 1) t = fmaxf(t, __shfl_xor(t, m, 32));
      float newM = fmaxf(Mst[v], t);
      float f  = exp2f(Mst[v] - newM);
      float p0 = exp2f(s01[0][v] - newM);
      float p1 = exp2f(s01[1][v] - newM);
      float rs = p0 + p1;
#pragma unroll
      for (int m = 1; m < 16; m <<= 1) rs += __shfl_xor(rs, m, 32);
      Lst[v] = Lst[v] * f + rs;
      Mst[v] = newM;
#pragma unroll
      for (int nf = 0; nf < 4; ++nf) acc[nf][v] *= f;
      // stash P (C layout -> LDS, k-major rows for the A-fragment reload)
      int mrow = v + 8 * half;
      Ps16[w * (16 * 2 * PS_STR) + mrow * (2 * PS_STR) + l16]      = f2bf(p0);
      Ps16[w * (16 * 2 * PS_STR) + mrow * (2 * PS_STR) + 16 + l16] = f2bf(p1);
    }

    // ---- O += P(16x32) @ V(32x64) ----
    FragAB pf;
#pragma unroll
    for (int v = 0; v < 8; ++v)
      pf.u[v] = Ps[w * (16 * PS_STR) + l16 * PS_STR + dofs[v]];
#pragma unroll
    for (int nf = 0; nf < 4; ++nf) {
      FragAB vf;   // B frag: col n = d, k = key, sequential dwords
#pragma unroll
      for (int v = 0; v < 8; ++v)
        vf.u[v] = Vt[(nf * 16 + l16) * VT_STR + half * 8 + v];
      acc[nf] = __builtin_amdgcn_wmma_f32_16x16x32_bf16(
          false, pf.v, false, vf.v, (short)0, acc[nf], false, false);
    }
  }

  // ---- normalize and store attn as bf16 [B,S,1024], col = h*64 + d ----
#pragma unroll
  for (int v = 0; v < 8; ++v) {
    float inv = 1.0f / Lst[v];
    int qr = qt * 128 + w * 16 + v + 8 * half;
#pragma unroll
    for (int nf = 0; nf < 4; ++nf) {
      int col = h * DK + nf * 16 + l16;
      Ob[(size_t)(b * SEQ + qr) * DMODEL + col] = f2bf(acc[nf][v] * inv);
    }
  }
}

// ---------------------------------------------------------------------------
// Host launcher
// ---------------------------------------------------------------------------
extern "C" void kernel_launch(void* const* d_in, const int* in_sizes, int n_in,
                              void* d_out, int out_size, void* d_ws, size_t ws_size,
                              hipStream_t stream) {
  (void)in_sizes; (void)n_in; (void)out_size; (void)ws_size;

  const float* x  = (const float*)d_in[0];
  const float* Wq = (const float*)d_in[1];
  const float* Wk = (const float*)d_in[2];
  const float* Wv = (const float*)d_in[3];
  const float* Wo = (const float*)d_in[4];
  float* out = (float*)d_out;

  char* ws = (char*)d_ws;
  const size_t MB = (size_t)1 << 20;
  unsigned short* xb  = (unsigned short*)(ws + 0);
  unsigned short* wqb = (unsigned short*)(ws + 16 * MB);
  unsigned short* wkb = (unsigned short*)(ws + 18 * MB);
  unsigned short* wvb = (unsigned short*)(ws + 20 * MB);
  unsigned short* wob = (unsigned short*)(ws + 22 * MB);
  unsigned short* Qb  = (unsigned short*)(ws + 24 * MB);
  unsigned short* Kb  = (unsigned short*)(ws + 40 * MB);
  unsigned short* Vb  = (unsigned short*)(ws + 56 * MB);
  unsigned short* Ab  = (unsigned short*)(ws + 72 * MB);

  // 1) pack to bf16
  {
    int n = MROWS * DMODEL;                 // 8,388,608
    pack_bf16_kernel<<<(n + 255) / 256, 256, 0, stream>>>(x, xb, n);
    int nw = DMODEL * DMODEL;               // 1,048,576
    pack_bf16_kernel<<<(nw + 255) / 256, 256, 0, stream>>>(Wq, wqb, nw);
    pack_bf16_kernel<<<(nw + 255) / 256, 256, 0, stream>>>(Wk, wkb, nw);
    pack_bf16_kernel<<<(nw + 255) / 256, 256, 0, stream>>>(Wv, wvb, nw);
    pack_bf16_kernel<<<(nw + 255) / 256, 256, 0, stream>>>(Wo, wob, nw);
  }

  // 2) Q/K/V projections -> [B,H,S,64] bf16
  dim3 gg(DMODEL / 128, MROWS / 128);       // (8, 64)
  gemm_xwT_kernel<<<gg, 256, 0, stream>>>((const unsigned*)xb, (const unsigned*)wqb,
                                          Qb, nullptr, 0);
  gemm_xwT_kernel<<<gg, 256, 0, stream>>>((const unsigned*)xb, (const unsigned*)wkb,
                                          Kb, nullptr, 0);
  gemm_xwT_kernel<<<gg, 256, 0, stream>>>((const unsigned*)xb, (const unsigned*)wvb,
                                          Vb, nullptr, 0);

  // 3) causal flash attention -> [B,S,1024] bf16
  dim3 ga(SEQ / 128, NHEADS, BATCH);        // (16, 16, 4)
  flash_attn_kernel<<<ga, 256, 0, stream>>>((const unsigned*)Qb, (const unsigned*)Kb,
                                            (const unsigned*)Vb, Ab);

  // 4) output projection -> fp32 d_out
  gemm_xwT_kernel<<<gg, 256, 0, stream>>>((const unsigned*)Ab, (const unsigned*)wob,
                                          nullptr, out, 1);
}
